// MultiHeadedAttention_44117904064873
// MI455X (gfx1250) — matrix operand order repaired and checked by reading, not compiled, not run on hardware
//
#include <hip/hip_runtime.h>

#define D_MODEL 512
#define N_HEADS 8
#define HEAD_DIM 64
#define SEQ 2048
#define BATCH 8

typedef __bf16 bhalf;
typedef __attribute__((ext_vector_type(16))) __bf16 bf16x16;
typedef __attribute__((ext_vector_type(8)))  float  f32x8;

static __device__ __forceinline__ f32x8 wmma_bf16(bf16x16 a, bf16x16 b, f32x8 c) {
  // D = A(16x32 bf16) * B(32x16 bf16) + C(16x16 f32)
  return __builtin_amdgcn_wmma_f32_16x16x32_bf16(false, a, false, b, (short)0, c,
                                                 false, false);
}

static __device__ __forceinline__ f32x8 zero8() {
  f32x8 z;
#pragma unroll
  for (int i = 0; i < 8; ++i) z[i] = 0.0f;
  return z;
}

// --- CDNA5 async global->LDS copy (ASYNCcnt-tracked), with safe fallback ----
#if defined(__has_builtin)
#  if __has_builtin(__builtin_amdgcn_global_load_async_to_lds_b128)
#    define USE_ASYNC_COPY 1
#  endif
#endif
#ifndef USE_ASYNC_COPY
#  define USE_ASYNC_COPY 0
#endif

typedef int i32x4v __attribute__((vector_size(16)));
typedef __attribute__((address_space(1))) i32x4v as1_i32x4;
typedef __attribute__((address_space(3))) i32x4v as3_i32x4;

static __device__ __forceinline__ void copy16_g2l(void* lds, const void* gmem) {
#if USE_ASYNC_COPY
  // global_load_async_to_lds_b128: 16B/lane, VGPR-free, tracked by ASYNCcnt.
  __builtin_amdgcn_global_load_async_to_lds_b128(
      (as1_i32x4*)(uintptr_t)gmem, (as3_i32x4*)(uintptr_t)lds, 0, 0);
#else
  *(uint4*)lds = *(const uint4*)gmem;
#endif
}

static __device__ __forceinline__ void wait_async_copies() {
#if USE_ASYNC_COPY
  asm volatile("s_wait_asynccnt 0" ::: "memory");
#endif
}

// ---------------------------------------------------------------------------
// Kernel 1: pointwise (conv1d k=1) projection, fp32 -> bf16, head-split layout.
//   out[co,n] = (sum_ci W[co,ci] * X[b,ci,n] + bias[co]) * scale
//   MODE 0/1 (Q/K): out layout [B][H][seq][64]   (co = d*8 + h)
//   MODE 2   (V)  : out layout [B][H][64][seq]
// ---------------------------------------------------------------------------
template <int MODE>
__global__ __launch_bounds__(256) void qkv_proj_kernel(
    const float* __restrict__ W, const float* __restrict__ bias,
    const float* __restrict__ X, bhalf* __restrict__ out, float scale) {
  const int lane = threadIdx.x & 31;
  const int wave = threadIdx.x >> 5;
  const int hi   = (lane >> 4) & 1;   // lane half (wave32)
  const int ln   = lane & 15;
  const int b    = blockIdx.y;
  const int tid  = blockIdx.x * 8 + wave;
  const int co0  = (tid & 31) * 16;   // 512/16 = 32 row tiles
  const int n0   = (tid >> 5) * 16;   // 2048/16 = 128 col tiles

  const float* Xb = X + (size_t)b * D_MODEL * SEQ;
  f32x8 acc = zero8();

  for (int k0 = 0; k0 < D_MODEL; k0 += 32) {
    bf16x16 aw, xb;
    const int arow = co0 + ln;
#pragma unroll
    for (int i = 0; i < 16; ++i) {  // A layout: k = (i&7) + 16*(i>>3) + 8*hi
      int kk = (i & 7) + ((i >> 3) << 4) + hi * 8;
      aw[i] = (bhalf)(W[(size_t)arow * D_MODEL + k0 + kk] * scale);
    }
    const int bcol = n0 + ln;
#pragma unroll
    for (int i = 0; i < 16; ++i) {  // B layout: k = (i&7) + 16*(i>>3) + 8*hi, n = ln
      int kk = (i & 7) + ((i >> 3) << 4) + hi * 8;
      xb[i] = (bhalf)Xb[(size_t)(k0 + kk) * SEQ + bcol];
    }
    acc = wmma_bf16(aw, xb, acc);
  }

#pragma unroll
  for (int r = 0; r < 8; ++r) {  // C layout: row = r + 8*hi, col = ln
    int co = co0 + r + hi * 8;
    int n  = n0 + ln;
    float v = acc[r] + bias[co] * scale;
    int d = co >> 3, hh = co & 7;  // reshape(512 -> [64,8]): co = d*8 + h
    size_t off;
    if (MODE == 2) off = (((size_t)b * N_HEADS + hh) * HEAD_DIM + d) * SEQ + n;
    else           off = (((size_t)b * N_HEADS + hh) * SEQ + n) * HEAD_DIM + d;
    out[off] = (bhalf)v;
  }
}

// ---------------------------------------------------------------------------
// Kernel 2: flash attention. Block = 8 waves x 16 query rows = 128 rows of one
// (b,h). 64-key chunks stream through double-buffered LDS via async
// global->LDS copies; additive-bias masking; online softmax whose fixed costs
// (cross-lane reductions, accumulator rescale) are amortized over 64 keys.
// ---------------------------------------------------------------------------
__global__ __launch_bounds__(256) void flash_attn_kernel(
    const bhalf* __restrict__ Q, const bhalf* __restrict__ K,
    const bhalf* __restrict__ V, const float* __restrict__ qmask,
    const float* __restrict__ kvmask, bhalf* __restrict__ X) {
  __shared__ __attribute__((aligned(16))) bhalf lK[2][64 * HEAD_DIM];  // [m][d]
  __shared__ __attribute__((aligned(16))) bhalf lV[2][HEAD_DIM * 64];  // [d][m]
  __shared__ __attribute__((aligned(16))) bhalf lP[8][16 * 64];        // per-wave

  const int lane = threadIdx.x & 31;
  const int wave = threadIdx.x >> 5;
  const int hi   = (lane >> 4) & 1;
  const int ln   = lane & 15;
  const int h    = blockIdx.y;
  const int b    = blockIdx.z;
  const int n0   = blockIdx.x * 128 + wave * 16;
  const size_t head  = ((size_t)b * N_HEADS + h) * (size_t)SEQ * HEAD_DIM;
  const size_t headV = ((size_t)b * N_HEADS + h) * (size_t)HEAD_DIM * SEQ;

  // Stage one 64-key chunk: K tile 64x64 bf16 (8KB) + V tile 64x64 (8KB).
  // 512 x 16B chunks per tile -> 2 copies per thread per tile.
  const int st = threadIdx.x;
  auto stage = [&](int buf, int m0) {
#pragma unroll
    for (int j = 0; j < 2; ++j) {
      int c   = st + 256 * j;
      int row = c >> 3;            // 8 x 16B segments per 128B row
      int seg = (c & 7) * 8;       // bf16 offset
      copy16_g2l(&lK[buf][row * HEAD_DIM + seg],
                 K + head + (size_t)(m0 + row) * HEAD_DIM + seg);
      copy16_g2l(&lV[buf][row * 64 + seg],
                 V + headV + (size_t)row * SEQ + m0 + seg);
    }
  };

  // Q fragments for d=0..31 and d=32..63 (Q pre-scaled by 1/sqrt(64))
  bf16x16 qa0, qa1;
  {
    const bhalf* Qp = Q + head + (size_t)(n0 + ln) * HEAD_DIM;
#pragma unroll
    for (int i = 0; i < 16; ++i) {
      int d = (i & 7) + ((i >> 3) << 4) + hi * 8;
      qa0[i] = Qp[d];
      qa1[i] = Qp[32 + d];
    }
  }
  // Query-row mask as an additive bias (0 or -1e9), fixed per kernel.
  float qb[8];
#pragma unroll
  for (int r = 0; r < 8; ++r) {
    float q = qmask[(size_t)b * SEQ + n0 + r + hi * 8];
    qb[r] = (q > 0.0f) ? 0.0f : -1.0e9f;
  }

  float mi[8], li[8];
  f32x8 acc[4];
#pragma unroll
  for (int r = 0; r < 8; ++r) { mi[r] = -3.0e38f; li[r] = 0.0f; }
#pragma unroll
  for (int t = 0; t < 4; ++t) acc[t] = zero8();

  stage(0, 0);  // prologue: fill buffer 0
  int cur = 0;

  for (int m0 = 0; m0 < SEQ; m0 += 64) {
    wait_async_copies();   // this wave's in-flight copies have landed
    __syncthreads();       // everyone's copies landed; prev reads of cur^1 done
    if (m0 + 64 < SEQ) stage(cur ^ 1, m0 + 64);  // overlap next chunk

    const bhalf* cK = lK[cur];
    const bhalf* cV = lV[cur];

    // S[t] = Q . K^T for key columns m0 + t*16 .. +15   (8 WMMAs)
    f32x8 s[4];
#pragma unroll
    for (int t = 0; t < 4; ++t) {
      bf16x16 kb0, kb1;
      int mcol = t * 16 + ln;
#pragma unroll
      for (int i = 0; i < 16; ++i) {
        int kk = (i & 7) + ((i >> 3) << 4) + hi * 8;
        kb0[i] = cK[mcol * HEAD_DIM + kk];
        kb1[i] = cK[mcol * HEAD_DIM + 32 + kk];
      }
      f32x8 z = zero8();
      z    = wmma_bf16(qa0, kb0, z);
      s[t] = wmma_bf16(qa1, kb1, z);
    }

    // Key-column mask as additive bias, one value per (lane, tile).
    float kvb[4];
#pragma unroll
    for (int t = 0; t < 4; ++t) {
      float kv = kvmask[(size_t)b * SEQ + m0 + t * 16 + ln];
      kvb[t] = (kv > 0.0f) ? 0.0f : -1.0e9f;
    }

    float cm[8], rs[8];
#pragma unroll
    for (int r = 0; r < 8; ++r) {
#pragma unroll
      for (int t = 0; t < 4; ++t) s[t][r] += qb[r] + kvb[t];
      cm[r] = fmaxf(fmaxf(s[0][r], s[1][r]), fmaxf(s[2][r], s[3][r]));
    }
    // row max: reduce across the 16 lanes holding one row's columns
#pragma unroll
    for (int r = 0; r < 8; ++r) {
#pragma unroll
      for (int x = 1; x < 16; x <<= 1)
        cm[r] = fmaxf(cm[r], __shfl_xor(cm[r], x, 32));
    }
#pragma unroll
    for (int r = 0; r < 8; ++r) {
      float mnew  = fmaxf(mi[r], cm[r]);
      float alpha = __expf(mi[r] - mnew);
      int row = r + hi * 8;
      float rsum = 0.0f;
#pragma unroll
      for (int t = 0; t < 4; ++t) {
        float e = __expf(s[t][r] - mnew);
        lP[wave][row * 64 + t * 16 + ln] = (bhalf)e;  // C layout -> LDS
        rsum += e;
      }
      rs[r] = rsum;
      mi[r] = mnew;
      li[r] *= alpha;
#pragma unroll
      for (int t = 0; t < 4; ++t) acc[t][r] *= alpha;
    }
#pragma unroll
    for (int r = 0; r < 8; ++r) {
#pragma unroll
      for (int x = 1; x < 16; x <<= 1)
        rs[r] += __shfl_xor(rs[r], x, 32);
      li[r] += rs[r];
    }

    // P back in A layout (two 16x32 fragments); same-wave LDS is in-order.
#pragma unroll
    for (int c = 0; c < 2; ++c) {
      bf16x16 pa;
#pragma unroll
      for (int i = 0; i < 16; ++i) {
        int kk = (i & 7) + ((i >> 3) << 4) + hi * 8;
        pa[i] = lP[wave][ln * 64 + c * 32 + kk];
      }
      // O[dt] += P . V  (8 WMMAs per chunk across both halves)
#pragma unroll
      for (int dt = 0; dt < 4; ++dt) {
        bf16x16 vb;
        int dcol = dt * 16 + ln;
#pragma unroll
        for (int i = 0; i < 16; ++i)
          vb[i] = cV[dcol * 64 + c * 32 + (i & 7) + ((i >> 3) << 4) + hi * 8];
        acc[dt] = wmma_bf16(pa, vb, acc[dt]);
      }
    }
    cur ^= 1;
  }

  // Normalize and emit bf16 activations, layout [B][N][512], ci = d*8 + h.
#pragma unroll
  for (int r = 0; r < 8; ++r)
    li[r] = (li[r] > 0.0f) ? 1.0f / li[r] : 0.0f;
#pragma unroll
  for (int dt = 0; dt < 4; ++dt) {
    int d  = dt * 16 + ln;
    int ci = d * N_HEADS + h;
#pragma unroll
    for (int r = 0; r < 8; ++r) {
      int n = n0 + r + hi * 8;
      X[((size_t)b * SEQ + n) * D_MODEL + ci] = (bhalf)(acc[dt][r] * li[r]);
    }
  }
}

// ---------------------------------------------------------------------------
// Kernel 3: output projection, bf16 activations [B][N][512] -> fp32 [B][512][N]
// ---------------------------------------------------------------------------
__global__ __launch_bounds__(256) void out_proj_kernel(
    const float* __restrict__ Wm, const float* __restrict__ bm,
    const bhalf* __restrict__ X, float* __restrict__ out) {
  const int lane = threadIdx.x & 31;
  const int wave = threadIdx.x >> 5;
  const int hi   = (lane >> 4) & 1;
  const int ln   = lane & 15;
  const int b    = blockIdx.y;
  const int tid  = blockIdx.x * 8 + wave;
  const int co0  = (tid & 31) * 16;
  const int n0   = (tid >> 5) * 16;

  f32x8 acc = zero8();
  for (int k0 = 0; k0 < D_MODEL; k0 += 32) {
    bf16x16 aw, xb;
    const int arow = co0 + ln;
#pragma unroll
    for (int i = 0; i < 16; ++i) {
      int kk = (i & 7) + ((i >> 3) << 4) + hi * 8;
      aw[i] = (bhalf)Wm[(size_t)arow * D_MODEL + k0 + kk];
    }
    const int bcol = n0 + ln;
#pragma unroll
    for (int i = 0; i < 16; ++i)
      xb[i] = X[((size_t)b * SEQ + bcol) * D_MODEL + k0 + (i & 7) + ((i >> 3) << 4) + hi * 8];
    acc = wmma_bf16(aw, xb, acc);
  }
#pragma unroll
  for (int r = 0; r < 8; ++r) {
    int co = co0 + r + hi * 8;
    int n  = n0 + ln;
    out[((size_t)b * D_MODEL + co) * SEQ + n] = acc[r] + bm[co];
  }
}

// ---------------------------------------------------------------------------
extern "C" void kernel_launch(void* const* d_in, const int* in_sizes, int n_in,
                              void* d_out, int out_size, void* d_ws, size_t ws_size,
                              hipStream_t stream) {
  const float* query  = (const float*)d_in[0];
  const float* key    = (const float*)d_in[1];
  const float* value  = (const float*)d_in[2];
  const float* qmask  = (const float*)d_in[3];
  const float* kvmask = (const float*)d_in[4];
  const float* Wq = (const float*)d_in[5];
  const float* bq = (const float*)d_in[6];
  const float* Wk = (const float*)d_in[7];
  const float* bk = (const float*)d_in[8];
  const float* Wv = (const float*)d_in[9];
  const float* bv = (const float*)d_in[10];
  const float* Wm = (const float*)d_in[11];
  const float* bm = (const float*)d_in[12];
  float* out = (float*)d_out;

  // Workspace: Q, K, V, X each B*H*SEQ*64 = 8M bf16 = 16 MB  (64 MB total)
  const size_t elems = (size_t)BATCH * N_HEADS * SEQ * HEAD_DIM;
  bhalf* Qb = (bhalf*)d_ws;
  bhalf* Kb = Qb + elems;
  bhalf* Vb = Kb + elems;
  bhalf* Xb = Vb + elems;

  dim3 blk(256);
  dim3 gproj((D_MODEL / 16) * (SEQ / 16) / 8, BATCH, 1);  // 512 x 8 blocks
  // Q pre-scaled by 1/sqrt(head_dim) = 0.125 (folds the score scaling)
  qkv_proj_kernel<0><<<gproj, blk, 0, stream>>>(Wq, bq, query, Qb, 0.125f);
  qkv_proj_kernel<1><<<gproj, blk, 0, stream>>>(Wk, bk, key,   Kb, 1.0f);
  qkv_proj_kernel<2><<<gproj, blk, 0, stream>>>(Wv, bv, value, Vb, 1.0f);

  dim3 gattn(SEQ / 128, N_HEADS, BATCH);  // 16 x 8 x 8 blocks
  flash_attn_kernel<<<gattn, blk, 0, stream>>>(Qb, Kb, Vb, qmask, kvmask, Xb);

  out_proj_kernel<<<gproj, blk, 0, stream>>>(Wm, bm, Xb, out);
}
